// MultiQueryAttention_44057774523023
// MI455X (gfx1250) — matrix operand order repaired and checked by reading, not compiled, not run on hardware
//
#include <hip/hip_runtime.h>

// ---------------------------------------------------------------------------
// Feature probes (compile-safe: full fallback path exists)
// ---------------------------------------------------------------------------
#if defined(__has_builtin)
#  if __has_builtin(__builtin_amdgcn_global_load_async_to_lds_b128)
#    define HAVE_ASYNC_LDS 1
#  else
#    define HAVE_ASYNC_LDS 0
#  endif
#  if __has_builtin(__builtin_amdgcn_s_wait_asynccnt)
#    define HAVE_WAIT_ASYNC 1
#  else
#    define HAVE_WAIT_ASYNC 0
#  endif
#else
#  define HAVE_ASYNC_LDS 0
#  define HAVE_WAIT_ASYNC 0
#endif

// ---------------------------------------------------------------------------
// Types for CDNA5 WMMA (wave32): 16x16x32 bf16 -> f32
// ---------------------------------------------------------------------------
typedef __bf16 bf16_t;
typedef bf16_t v8bf  __attribute__((ext_vector_type(8)));
typedef bf16_t v16bf __attribute__((ext_vector_type(16)));
typedef float  v8f   __attribute__((ext_vector_type(8)));

union FragBF16 {
  v16bf v;
  v8bf  h[2];
};

__device__ __forceinline__ v8f wmma_bf16(const v16bf& a, const v16bf& b, const v8f& c) {
  return __builtin_amdgcn_wmma_f32_16x16x32_bf16(
      /*neg_a=*/false, a, /*neg_b=*/false, b,
      /*c_mod=*/(short)0, c, /*reuse_a=*/false, /*reuse_b=*/false);
}

__device__ __forceinline__ unsigned short f32_to_bf16(float f) {
  unsigned int u = __float_as_uint(f);
  u += 0x7FFFu + ((u >> 16) & 1u);  // round to nearest even
  return (unsigned short)(u >> 16);
}

__device__ __forceinline__ void store_out(float* p, float v) { *p = v; }
__device__ __forceinline__ void store_out(unsigned short* p, float v) { *p = f32_to_bf16(v); }

// ---------------------------------------------------------------------------
// Async global -> LDS copy helpers (CDNA5 GLOBAL_LOAD_ASYNC_TO_LDS_B128).
// Builtin prototype (from clang diagnostic): param0 = v4i in addrspace(1),
// (global source), param1 = LDS destination, then imm offset + cpol.
// ---------------------------------------------------------------------------
typedef int v4i __attribute__((vector_size(16)));
#if HAVE_ASYNC_LDS
typedef __attribute__((address_space(1))) v4i* gv4i_ptr;
typedef __attribute__((address_space(3))) v4i* lv4i_ptr;
#endif

__device__ __forceinline__ void async_copy_b128(void* lds_dst, const void* g_src) {
#if HAVE_ASYNC_LDS
  __builtin_amdgcn_global_load_async_to_lds_b128(
      (gv4i_ptr)(g_src), (lv4i_ptr)(lds_dst), 0, 0);
#else
  *(uint4*)lds_dst = *(const uint4*)g_src;
#endif
}

__device__ __forceinline__ void wait_async_zero() {
#if HAVE_ASYNC_LDS
#  if HAVE_WAIT_ASYNC
  __builtin_amdgcn_s_wait_asynccnt(0);
#  else
  asm volatile("s_wait_asynccnt 0x0" ::: "memory");
#  endif
#endif
}

// ---------------------------------------------------------------------------
// fp32 -> bf16 conversion (n must be a multiple of 4)
// ---------------------------------------------------------------------------
__global__ __launch_bounds__(256) void cvt_f32_bf16_kernel(
    const float* __restrict__ in, unsigned short* __restrict__ out, int n) {
  int i = (blockIdx.x * blockDim.x + threadIdx.x) * 4;
  if (i < n) {
    float4 f = *(const float4*)(in + i);
    uint2 o;
    o.x = (unsigned int)f32_to_bf16(f.x) | ((unsigned int)f32_to_bf16(f.y) << 16);
    o.y = (unsigned int)f32_to_bf16(f.z) | ((unsigned int)f32_to_bf16(f.w) << 16);
    *(uint2*)(out + i) = o;
  }
}

// ---------------------------------------------------------------------------
// Tiled bf16 WMMA GEMM:  C[M,N] = (A[M,K] * W[K,N] + bias[N]) * alpha
//   Block: 256 threads = 8 waves, tile 128x128, BK=64 (two WMMA k-steps),
//   LDS double-buffered. A staged via async-to-LDS (or register path),
//   W staged transposed ([n][k]) through VGPR k-pair packing so that B
//   fragments are two contiguous 8-half reads. One barrier per BK stage; the
//   next stage's global traffic overlaps with the 16 WMMAs of this stage.
// ---------------------------------------------------------------------------
template <typename OutT>
__global__ __launch_bounds__(256) void gemm_bf16_wmma(
    const unsigned short* __restrict__ A,   // [M,K] bf16 bits, row-major
    const unsigned short* __restrict__ W,   // [K,N] bf16 bits, row-major
    const float* __restrict__ bias,         // [N]
    OutT* __restrict__ C,                   // [M,N]
    int M, int N, int K, float alpha) {
  constexpr int BM = 128, BN = 128, BK = 64;
  __shared__ __attribute__((aligned(16))) unsigned short sA[2][BM][BK + 8];  // 144B rows
  __shared__ __attribute__((aligned(16))) unsigned short sB[2][BN][BK + 8];

  const int tid   = threadIdx.x;
  const int lane  = tid & 31;
  const int wave  = tid >> 5;
  const int waveM = wave >> 2;      // 0..1
  const int waveN = wave & 3;       // 0..3
  const int blockM = blockIdx.y * BM;
  const int blockN = blockIdx.x * BN;
  const int r16  = lane & 15;
  const int hsel = (lane >> 4) * 8;
  const int ksel = (lane >> 4) * 16;

  // Per-thread staging coordinates.
  // A: 4 chunks of 8 halves: c = tid + i*256 -> row = c>>3, col8 = (c&7)*8
  // B: 2 pack-chunks (2 k-rows x 8 n): c = tid + i*256 -> kk = (c>>4)*2, n8 = (c&15)*8

  v8f acc[4][2];
#pragma unroll
  for (int mi = 0; mi < 4; ++mi)
#pragma unroll
    for (int ni = 0; ni < 2; ++ni)
#pragma unroll
      for (int j = 0; j < 8; ++j) acc[mi][ni][j] = 0.0f;

  auto pack_store_b = [&](unsigned short(*sb)[BK + 8], int n8, int kk,
                          const uint4& r0, const uint4& r1) {
    const unsigned short* p0 = (const unsigned short*)&r0;
    const unsigned short* p1 = (const unsigned short*)&r1;
#pragma unroll
    for (int i = 0; i < 8; ++i)
      *(unsigned int*)&sb[n8 + i][kk] =
          (unsigned int)p0[i] | ((unsigned int)p1[i] << 16);
  };

  // ---------------- prologue: stage k0 = 0 into buffer 0 ----------------
#pragma unroll
  for (int i = 0; i < 4; ++i) {
    int c = tid + i * 256, row = c >> 3, col8 = (c & 7) * 8;
    async_copy_b128(&sA[0][row][col8], A + (size_t)(blockM + row) * K + col8);
  }
#pragma unroll
  for (int i = 0; i < 2; ++i) {
    int c = tid + i * 256, kk = (c >> 4) * 2, n8 = (c & 15) * 8;
    uint4 r0 = *(const uint4*)(W + (size_t)kk * N + blockN + n8);
    uint4 r1 = *(const uint4*)(W + (size_t)(kk + 1) * N + blockN + n8);
    pack_store_b(sB[0], n8, kk, r0, r1);
  }
  wait_async_zero();

  int buf = 0;
  for (int k0 = 0; k0 < K; k0 += BK) {
    __syncthreads();  // staging of `buf` visible; all reads of buf^1 retired
    const bool next = (k0 + BK) < K;

    uint4 bR0[2], bR1[2];
#if !HAVE_ASYNC_LDS
    uint4 aR[4];
#endif
    if (next) {
      // Issue next-stage global traffic now; it overlaps with the WMMAs below.
#if HAVE_ASYNC_LDS
#pragma unroll
      for (int i = 0; i < 4; ++i) {
        int c = tid + i * 256, row = c >> 3, col8 = (c & 7) * 8;
        async_copy_b128(&sA[buf ^ 1][row][col8],
                        A + (size_t)(blockM + row) * K + k0 + BK + col8);
      }
#else
#pragma unroll
      for (int i = 0; i < 4; ++i) {
        int c = tid + i * 256, row = c >> 3, col8 = (c & 7) * 8;
        aR[i] = *(const uint4*)(A + (size_t)(blockM + row) * K + k0 + BK + col8);
      }
#endif
#pragma unroll
      for (int i = 0; i < 2; ++i) {
        int c = tid + i * 256, kk = (c >> 4) * 2, n8 = (c & 15) * 8;
        bR0[i] = *(const uint4*)(W + (size_t)(k0 + BK + kk) * N + blockN + n8);
        bR1[i] = *(const uint4*)(W + (size_t)(k0 + BK + kk + 1) * N + blockN + n8);
      }
    }

    // ---- compute: 2 k-steps x (4x2) WMMA tiles ----
#pragma unroll
    for (int ks = 0; ks < BK; ks += 32) {
      FragBF16 af[4], bfr[2];
#pragma unroll
      for (int mi = 0; mi < 4; ++mi) {
        int row = waveM * 64 + mi * 16 + r16;
        af[mi].h[0] = *(const v8bf*)&sA[buf][row][ks + hsel];
        af[mi].h[1] = *(const v8bf*)&sA[buf][row][ks + 16 + hsel];
      }
#pragma unroll
      for (int ni = 0; ni < 2; ++ni) {
        int col = waveN * 32 + ni * 16 + r16;
        bfr[ni].h[0] = *(const v8bf*)&sB[buf][col][ks + hsel];
        bfr[ni].h[1] = *(const v8bf*)&sB[buf][col][ks + 16 + hsel];
      }
#pragma unroll
      for (int mi = 0; mi < 4; ++mi)
#pragma unroll
        for (int ni = 0; ni < 2; ++ni)
          acc[mi][ni] = wmma_bf16(af[mi].v, bfr[ni].v, acc[mi][ni]);
    }

    if (next) {
#if !HAVE_ASYNC_LDS
#pragma unroll
      for (int i = 0; i < 4; ++i) {
        int c = tid + i * 256, row = c >> 3, col8 = (c & 7) * 8;
        *(uint4*)&sA[buf ^ 1][row][col8] = aR[i];
      }
#endif
#pragma unroll
      for (int i = 0; i < 2; ++i) {
        int c = tid + i * 256, kk = (c >> 4) * 2, n8 = (c & 15) * 8;
        pack_store_b(sB[buf ^ 1], n8, kk, bR0[i], bR1[i]);
      }
      wait_async_zero();  // this stage's async copies must land before next barrier
    }
    buf ^= 1;
  }

  // ---- epilogue: C = (acc + bias) * alpha ----
  const int rowoff = (lane >> 4) * 8;
#pragma unroll
  for (int ni = 0; ni < 2; ++ni) {
    int col  = blockN + waveN * 32 + ni * 16 + r16;
    float bv = bias[col];
#pragma unroll
    for (int mi = 0; mi < 4; ++mi) {
#pragma unroll
      for (int j = 0; j < 8; ++j) {
        int row = blockM + waveM * 64 + mi * 16 + rowoff + j;
        store_out(C + (size_t)row * N + col, (acc[mi][ni][j] + bv) * alpha);
      }
    }
  }
}

// ---------------------------------------------------------------------------
// MQA flash attention.
//   Grid: (S/64, H, B), block 128 threads = 4 waves. Wave w owns 16 query rows.
//   Q (pre-scaled by 1/sqrt(Hd)) fragments live in registers across the whole
//   key loop. K chunk staged row-major via async-to-LDS (B-frag of Q*K^T is
//   contiguous); V chunk staged transposed. Online softmax in WMMA C layout.
// ---------------------------------------------------------------------------
__global__ __launch_bounds__(128) void mqa_attention_kernel(
    const unsigned short* __restrict__ Qs,  // [B,S,D] bf16, scaled
    const unsigned short* __restrict__ Kp,  // [B,S,Hd] bf16
    const unsigned short* __restrict__ Vp,  // [B,S,Hd] bf16
    unsigned short* __restrict__ ctx,       // [B,S,D] bf16
    int S) {
  constexpr int Hd = 128, D = 2048, BM = 64, BN = 64;
  const int b = blockIdx.z, h = blockIdx.y, mt = blockIdx.x;
  const int tid = threadIdx.x, lane = tid & 31, wave = tid >> 5;
  const int r16 = lane & 15;
  const int hsel = (lane >> 4) * 8;
  const int rowoff = (lane >> 4) * 8;

  __shared__ __attribute__((aligned(16))) unsigned short sK[BN][Hd + 8];
  __shared__ __attribute__((aligned(16))) unsigned short sVt[Hd][BN + 8];
  __shared__ __attribute__((aligned(16))) unsigned short sP[4][16][BN + 8];

  // ---- Q fragments: 4 x (16x32), straight from global ----
  const unsigned short* Qbase =
      Qs + ((size_t)b * S + (size_t)mt * BM + wave * 16) * D + h * Hd;
  FragBF16 qf[4];
#pragma unroll
  for (int kt = 0; kt < 4; ++kt) {
    const unsigned short* qrow = Qbase + (size_t)r16 * D + kt * 32;
    qf[kt].h[0] = *(const v8bf*)(qrow + hsel);
    qf[kt].h[1] = *(const v8bf*)(qrow + 16 + hsel);
  }

  float m_run[8], l_run[8];
  v8f o_acc[8];
#pragma unroll
  for (int j = 0; j < 8; ++j) { m_run[j] = -3.0e38f; l_run[j] = 0.0f; }
#pragma unroll
  for (int t = 0; t < 8; ++t)
#pragma unroll
    for (int j = 0; j < 8; ++j) o_acc[t][j] = 0.0f;

  for (int n0 = 0; n0 < S; n0 += BN) {
    // ---- stage K chunk: 64 x 128 halves (async direct to LDS) ----
    const unsigned short* Kc = Kp + ((size_t)b * S + n0) * Hd;
#pragma unroll
    for (int i = 0; i < 8; ++i) {
      int c    = tid + i * 128;
      int row  = c >> 4;
      int col8 = (c & 15) * 8;
      async_copy_b128(&sK[row][col8], Kc + (size_t)row * Hd + col8);
    }
    // ---- stage V chunk transposed (VGPR k-pair packing) ----
    const unsigned short* Vc = Vp + ((size_t)b * S + n0) * Hd;
#pragma unroll
    for (int i = 0; i < 4; ++i) {
      int c  = tid + i * 128;
      int kk = (c >> 4) * 2;
      int n8 = (c & 15) * 8;
      uint4 r0 = *(const uint4*)(Vc + (size_t)kk * Hd + n8);
      uint4 r1 = *(const uint4*)(Vc + (size_t)(kk + 1) * Hd + n8);
      const unsigned short* p0 = (const unsigned short*)&r0;
      const unsigned short* p1 = (const unsigned short*)&r1;
#pragma unroll
      for (int j = 0; j < 8; ++j)
        *(unsigned int*)&sVt[n8 + j][kk] =
            (unsigned int)p0[j] | ((unsigned int)p1[j] << 16);
    }
    wait_async_zero();
    __syncthreads();

    // ---- scores: S16x64 = Q(16x128) * K^T, 4 col-tiles x 4 k-steps ----
    v8f sacc[4];
#pragma unroll
    for (int nt = 0; nt < 4; ++nt) {
#pragma unroll
      for (int j = 0; j < 8; ++j) sacc[nt][j] = 0.0f;
#pragma unroll
      for (int kt = 0; kt < 4; ++kt) {
        int kb = kt * 32 + (lane >> 4) * 8;
        FragBF16 kf;
        kf.h[0] = *(const v8bf*)&sK[nt * 16 + r16][kb];
        kf.h[1] = *(const v8bf*)&sK[nt * 16 + r16][kb + 16];
        sacc[nt] = wmma_bf16(qf[kt].v, kf.v, sacc[nt]);
      }
    }

    // ---- online softmax ----
    float p_scale[8];
#pragma unroll
    for (int j = 0; j < 8; ++j) {
      float v = fmaxf(fmaxf(sacc[0][j], sacc[1][j]), fmaxf(sacc[2][j], sacc[3][j]));
#pragma unroll
      for (int off = 1; off < 16; off <<= 1) v = fmaxf(v, __shfl_xor(v, off, 32));
      float mn   = fmaxf(m_run[j], v);
      p_scale[j] = __expf(m_run[j] - mn);
      m_run[j]   = mn;
    }
    float rs[8];
#pragma unroll
    for (int j = 0; j < 8; ++j) rs[j] = 0.0f;
#pragma unroll
    for (int nt = 0; nt < 4; ++nt)
#pragma unroll
      for (int j = 0; j < 8; ++j) {
        float p = __expf(sacc[nt][j] - m_run[j]);
        rs[j] += p;
        sP[wave][j + rowoff][nt * 16 + r16] = f32_to_bf16(p);
      }
#pragma unroll
    for (int j = 0; j < 8; ++j) {
      float v = rs[j];
#pragma unroll
      for (int off = 1; off < 16; off <<= 1) v += __shfl_xor(v, off, 32);
      l_run[j] = l_run[j] * p_scale[j] + v;
    }
#pragma unroll
    for (int t = 0; t < 8; ++t)
#pragma unroll
      for (int j = 0; j < 8; ++j) o_acc[t][j] *= p_scale[j];

    // ---- O += P(16x64) * V(64x128): 8 col-tiles x 2 k-steps ----
    FragBF16 pf[2];
#pragma unroll
    for (int kt = 0; kt < 2; ++kt) {
      const unsigned short* prow = &sP[wave][r16][kt * 32];
      pf[kt].h[0] = *(const v8bf*)(prow + hsel);
      pf[kt].h[1] = *(const v8bf*)(prow + 16 + hsel);
    }
#pragma unroll
    for (int t = 0; t < 8; ++t) {
#pragma unroll
      for (int kt = 0; kt < 2; ++kt) {
        int kb = kt * 32 + (lane >> 4) * 8;
        FragBF16 vf;
        vf.h[0] = *(const v8bf*)&sVt[t * 16 + r16][kb];
        vf.h[1] = *(const v8bf*)&sVt[t * 16 + r16][kb + 16];
        o_acc[t] = wmma_bf16(pf[kt].v, vf.v, o_acc[t]);
      }
    }
    __syncthreads();
  }

  // ---- finalize: ctx[b, row, h*Hd + col] = O / l ----
  unsigned short* obase =
      ctx + ((size_t)b * S + (size_t)mt * BM + wave * 16) * D + h * Hd;
#pragma unroll
  for (int j = 0; j < 8; ++j) {
    float inv = 1.0f / l_run[j];
    int row = j + rowoff;
#pragma unroll
    for (int t = 0; t < 8; ++t)
      obase[(size_t)row * D + t * 16 + r16] = f32_to_bf16(o_acc[t][j] * inv);
  }
}

// ---------------------------------------------------------------------------
// Host-side orchestration
// ---------------------------------------------------------------------------
extern "C" void kernel_launch(void* const* d_in, const int* in_sizes, int n_in,
                              void* d_out, int out_size, void* d_ws, size_t ws_size,
                              hipStream_t stream) {
  (void)in_sizes; (void)n_in; (void)out_size; (void)ws_size;
  constexpr int B = 2, S = 2048, D = 2048, H = 16, Hd = 128;
  const int M = B * S;                   // 4096
  const size_t E   = (size_t)B * S * D;  // 8,388,608
  const size_t W2  = (size_t)D * D;      // 4,194,304
  const size_t Wsm = (size_t)D * Hd;     // 262,144
  const size_t SHd = (size_t)B * S * Hd; // 524,288

  const float* query = (const float*)d_in[0];
  const float* key_  = (const float*)d_in[1];
  const float* value = (const float*)d_in[2];
  const float* Wq = (const float*)d_in[3];
  const float* bq = (const float*)d_in[4];
  const float* Wk = (const float*)d_in[5];
  const float* bk = (const float*)d_in[6];
  const float* Wv = (const float*)d_in[7];
  const float* bv = (const float*)d_in[8];
  const float* Wo = (const float*)d_in[9];
  const float* bo = (const float*)d_in[10];
  float* out = (float*)d_out;

  char* ws = (char*)d_ws;
  size_t off = 0;
  auto alloc16 = [&](size_t elems) {
    unsigned short* p = (unsigned short*)(ws + off);
    off = (off + elems * 2 + 255) & ~(size_t)255;
    return p;
  };
  unsigned short* qh  = alloc16(E);
  unsigned short* kh  = alloc16(E);
  unsigned short* vh  = alloc16(E);
  unsigned short* wqh = alloc16(W2);
  unsigned short* wkh = alloc16(Wsm);
  unsigned short* wvh = alloc16(Wsm);
  unsigned short* woh = alloc16(W2);
  unsigned short* Qsc = alloc16(E);    // scaled Q projection
  unsigned short* Kp  = alloc16(SHd);  // K projection
  unsigned short* Vp  = alloc16(SHd);  // V projection
  unsigned short* ctx = alloc16(E);    // attention output, [B,S,H*Hd]

  auto cvt = [&](const float* src, unsigned short* dst, size_t n) {
    int blocks = (int)((n / 4 + 255) / 256);
    cvt_f32_bf16_kernel<<<blocks, 256, 0, stream>>>(src, dst, (int)n);
  };
  cvt(query, qh, E);
  cvt(key_,  kh, E);
  cvt(value, vh, E);
  cvt(Wq, wqh, W2);
  cvt(Wk, wkh, Wsm);
  cvt(Wv, wvh, Wsm);
  cvt(Wo, woh, W2);

  const float scale = 0.08838834764831845f;  // 1/sqrt(128)

  // Q projection (scale folded in), bf16 output
  gemm_bf16_wmma<unsigned short><<<dim3(D / 128, M / 128), 256, 0, stream>>>(
      qh, wqh, bq, Qsc, M, D, D, scale);
  // K projection, bf16 output
  gemm_bf16_wmma<unsigned short><<<dim3(Hd / 128, M / 128), 256, 0, stream>>>(
      kh, wkh, bk, Kp, M, Hd, D, 1.0f);
  // V projection, bf16 output
  gemm_bf16_wmma<unsigned short><<<dim3(Hd / 128, M / 128), 256, 0, stream>>>(
      vh, wvh, bv, Vp, M, Hd, D, 1.0f);

  // MQA flash attention
  mqa_attention_kernel<<<dim3(S / 64, H, B), 128, 0, stream>>>(Qsc, Kp, Vp, ctx, S);

  // Output projection, fp32 output straight into d_out
  gemm_bf16_wmma<float><<<dim3(D / 128, M / 128), 256, 0, stream>>>(
      ctx, woh, bo, out, M, D, D, 1.0f);
}
